// SelfAttention_51831665328247
// MI455X (gfx1250) — matrix operand order repaired and checked by reading, not compiled, not run on hardware
//
#include <hip/hip_runtime.h>

// ---------------------------------------------------------------------------
// MHA forward for MI455X (gfx1250): bf16 WMMA panel GEMMs + fused flash attn.
// All matrix math goes through v_wmma_f32_16x16x32_bf16.
// GEMM uses 2x4 register blocking: 8 WMMAs per 6 fragment loads per k-step.
// ---------------------------------------------------------------------------

typedef __attribute__((ext_vector_type(16))) __bf16 bf16x16;
typedef __attribute__((ext_vector_type(8)))  float  f32x8;

constexpr int kSeq   = 2048;
constexpr int kBatch = 2;
constexpr int kHeads = 16;
constexpr int kHd    = 64;
constexpr int kHid   = 1024;
constexpr int kRows  = kBatch * kSeq;   // 4096
constexpr int kMT    = kRows / 16;      // 256 M-tiles
constexpr int kNT    = kHid / 16;       // 64  N-tiles
constexpr int kKS    = kHid / 32;       // 32  K-steps of 32

static __device__ __forceinline__
f32x8 wmma_bf16(bf16x16 a, bf16x16 b, f32x8 c) {
  return __builtin_amdgcn_wmma_f32_16x16x32_bf16(
      /*neg_a=*/false, a, /*neg_b=*/false, b,
      /*c_mod=*/(short)0, c, /*reuse_a=*/false, /*reuse_b=*/false);
}

// A-layout (16x32 bf16, ISA 7.12.2): element i of lane l covers
//   k = (i&7) + (i>=8 ? 16:0) + (l>=16 ? 8:0), row m = l&15.
// Inverse: laneA = m + 16*((koff>>3)&1); iA = (koff&7) + ((koff&16)?8:0).
// B-layout (32x16 bf16): lane l holds n = l&15, k(i) = (i&7) + (i>=8 ? 16:0) + (l>=16 ? 8:0), as A.

// ---- pack x (fp32 row-major MxK) into A panels [mtile][kstep][lane][16] ----
__global__ __launch_bounds__(256)
void pack_A_f32(const float* __restrict__ src, __bf16* __restrict__ dst,
                int cols) {
  int idx    = blockIdx.x * blockDim.x + threadIdx.x;
  int ksteps = cols >> 5;
  int i      = idx & 15;
  int lane   = (idx >> 4) & 31;
  int kstep  = (idx >> 9) % ksteps;
  int mtile  = idx / (512 * ksteps);
  int m      = mtile * 16 + (lane & 15);
  int koff   = (i & 7) + ((i & 8) ? 16 : 0) + ((lane & 16) ? 8 : 0);
  int k      = kstep * 32 + koff;
  dst[idx] = (__bf16)src[(size_t)m * cols + k];
}

// ---- pack W (fp32 row-major KxN) into B panels [ntile][kstep][lane][16] ----
__global__ __launch_bounds__(256)
void pack_B_f32(const float* __restrict__ src, __bf16* __restrict__ dst,
                int K, int N) {
  int idx    = blockIdx.x * blockDim.x + threadIdx.x;
  int ksteps = K >> 5;
  int i      = idx & 15;
  int lane   = (idx >> 4) & 31;
  int kstep  = (idx >> 9) % ksteps;
  int ntile  = idx / (512 * ksteps);
  int n      = ntile * 16 + (lane & 15);
  int k      = kstep * 32 + (i & 7) + ((i & 8) ? 16 : 0) + ((lane & 16) ? 8 : 0);
  dst[idx] = (__bf16)src[(size_t)k * N + n];
}

// ---- epilogue scatter for one 16x16 C tile ---------------------------------
// mode 0: -> Q A-panels   [bh][qtile(128)][kstep(2)][lane][16]  (x 1/8 scale)
// mode 1: -> K B-panels   [bh][keytile(128)][kstep(2)][lane][16]  (n=key,k=d)
// mode 2: -> V B-panels   [bh][dtile(4)][kvstep(64)][lane][16]    (n=d,k=key)
// mode 3: -> fp32 row-major out [4096][1024]
static __device__ __forceinline__
void epilogue_tile(f32x8 acc, int mtile, int ntile, int lane,
                   const float* __restrict__ bias, void* __restrict__ outp,
                   int mode) {
  int hi = (lane >> 4) & 1;
  int ln = lane & 15;
  int n  = ntile * 16 + ln;
  float bn = bias[n];

#pragma unroll
  for (int r = 0; r < 8; ++r) {
    float v = acc[r] + bn;
    int m = mtile * 16 + r + 8 * hi;
    if (mode == 3) {
      ((float*)outp)[(size_t)m * kHid + n] = v;
    } else {
      int bidx = m >> 11;          // batch
      int s    = m & (kSeq - 1);   // sequence position
      int h    = n >> 6;           // head
      int d    = n & (kHd - 1);    // within-head dim
      int bh   = bidx * kHeads + h;
      __bf16* dst = (__bf16*)outp;
      if (mode == 0) {             // Q, A-layout, fold softmax scale 1/sqrt(64)
        int qtile = s >> 4, kst = d >> 5, koff = d & 31;
        int laneA = (s & 15) + 16 * ((koff >> 3) & 1);
        int iA    = (koff & 7) + ((koff & 16) ? 8 : 0);
        dst[((((size_t)bh * 128 + qtile) * 2 + kst) * 32 + laneA) * 16 + iA] =
            (__bf16)(v * 0.125f);
      } else if (mode == 1) {      // K, B-layout: n=key, k=d
        int ntK = s >> 4, kst = d >> 5, koff = d & 31;
        int laneB = (s & 15) + 16 * ((koff >> 3) & 1);
        int iB    = (koff & 7) + ((koff & 16) ? 8 : 0);
        dst[((((size_t)bh * 128 + ntK) * 2 + kst) * 32 + laneB) * 16 + iB] =
            (__bf16)v;
      } else {                     // V, B-layout: n=d, k=key
        int ntV = d >> 4, kst = s >> 5, koff = s & 31;
        int laneB = (d & 15) + 16 * ((koff >> 3) & 1);
        int iB    = (koff & 7) + ((koff & 16) ? 8 : 0);
        dst[((((size_t)bh * 4 + ntV) * 64 + kst) * 32 + laneB) * 16 + iB] =
            (__bf16)v;
      }
    }
  }
}

// ---- panel GEMM: each wave computes a 32x64 macro-tile (2x4 WMMA tiles) ----
__global__ __launch_bounds__(256)
void gemm_wmma(const __bf16* __restrict__ Apan,
               const __bf16* __restrict__ Bpan,
               const float*  __restrict__ bias,
               void* __restrict__ outp, int mode) {
  int wave   = threadIdx.x >> 5;
  int lane   = threadIdx.x & 31;
  int wl     = blockIdx.x * 8 + wave;   // 2048 wave-tiles total
  int ntg    = wl & 15;                 // 16 n-groups of 4 tiles
  int mtg    = wl >> 4;                 // 128 m-groups of 2 tiles
  int mtile0 = mtg * 2;
  int ntile0 = ntg * 4;

  const bf16x16* Ap0 = (const bf16x16*)Apan + (size_t)mtile0 * kKS * 32 + lane;
  const bf16x16* Ap1 = Ap0 + (size_t)kKS * 32;
  const bf16x16* Bp  = (const bf16x16*)Bpan + (size_t)ntile0 * kKS * 32 + lane;

  f32x8 acc[2][4] = {};
#pragma unroll 2
  for (int ks = 0; ks < kKS; ++ks) {
    __builtin_prefetch(Bp + (ks + 8) * 32, 0, 1);   // branch-free speculative
    bf16x16 a0 = Ap0[ks * 32];
    bf16x16 a1 = Ap1[ks * 32];
    bf16x16 b0 = Bp[ks * 32 + 0 * kKS * 32];
    bf16x16 b1 = Bp[ks * 32 + 1 * kKS * 32];
    bf16x16 b2 = Bp[ks * 32 + 2 * kKS * 32];
    bf16x16 b3 = Bp[ks * 32 + 3 * kKS * 32];
    acc[0][0] = wmma_bf16(a0, b0, acc[0][0]);
    acc[0][1] = wmma_bf16(a0, b1, acc[0][1]);
    acc[0][2] = wmma_bf16(a0, b2, acc[0][2]);
    acc[0][3] = wmma_bf16(a0, b3, acc[0][3]);
    acc[1][0] = wmma_bf16(a1, b0, acc[1][0]);
    acc[1][1] = wmma_bf16(a1, b1, acc[1][1]);
    acc[1][2] = wmma_bf16(a1, b2, acc[1][2]);
    acc[1][3] = wmma_bf16(a1, b3, acc[1][3]);
  }

#pragma unroll
  for (int mi = 0; mi < 2; ++mi)
#pragma unroll
    for (int ni = 0; ni < 4; ++ni)
      epilogue_tile(acc[mi][ni], mtile0 + mi, ntile0 + ni, lane, bias, outp,
                    mode);
}

// ---- fused flash attention: one wave per 16-query tile per (b,h) ----------
__global__ __launch_bounds__(256)
void flash_attn(const __bf16* __restrict__ qA, const __bf16* __restrict__ kB,
                const __bf16* __restrict__ vB, __bf16* __restrict__ ctxA) {
  __shared__ __bf16 plds[8][16 * 32];   // per-wave 16x32 P staging (1 KB each)

  int wave  = threadIdx.x >> 5;
  int lane  = threadIdx.x & 31;
  int bh    = blockIdx.y;
  int qtile = blockIdx.x * 8 + wave;
  int hi    = (lane >> 4) & 1;
  int ln    = lane & 15;

  const bf16x16* Qp = (const bf16x16*)qA + ((size_t)bh * 128 + qtile) * 64 + lane;
  bf16x16 qa0 = Qp[0];     // d 0..31   (scale pre-folded)
  bf16x16 qa1 = Qp[32];    // d 32..63

  const bf16x16* Kbase = (const bf16x16*)kB + (size_t)bh * 128 * 64 + lane;
  const bf16x16* Vbase = (const bf16x16*)vB + (size_t)bh * 4 * 64 * 32 + lane;
  __bf16* P = &plds[wave][0];

  f32x8 O[4] = {};
  float mrun[8], lrun[8];
#pragma unroll
  for (int r = 0; r < 8; ++r) { mrun[r] = -3.0e38f; lrun[r] = 0.0f; }

  for (int kv = 0; kv < 64; ++kv) {            // 32 keys per step
    const bf16x16* Kt0 = Kbase + (size_t)(kv * 2 + 0) * 64;
    const bf16x16* Kt1 = Kbase + (size_t)(kv * 2 + 1) * 64;
    f32x8 s0 = {}, s1 = {};
    s0 = wmma_bf16(qa0, Kt0[0],  s0);
    s0 = wmma_bf16(qa1, Kt0[32], s0);
    s1 = wmma_bf16(qa0, Kt1[0],  s1);
    s1 = wmma_bf16(qa1, Kt1[32], s1);

    float scl[8];
#pragma unroll
    for (int r = 0; r < 8; ++r) {
      float t = fmaxf(s0[r], s1[r]);
      t = fmaxf(t, __shfl_xor(t, 1));
      t = fmaxf(t, __shfl_xor(t, 2));
      t = fmaxf(t, __shfl_xor(t, 4));
      t = fmaxf(t, __shfl_xor(t, 8));
      float mnew = fmaxf(mrun[r], t);
      float p0 = __expf(s0[r] - mnew);
      float p1 = __expf(s1[r] - mnew);
      s0[r] = p0; s1[r] = p1;
      float rs = p0 + p1;
      rs += __shfl_xor(rs, 1);
      rs += __shfl_xor(rs, 2);
      rs += __shfl_xor(rs, 4);
      rs += __shfl_xor(rs, 8);
      scl[r]  = __expf(mrun[r] - mnew);
      lrun[r] = lrun[r] * scl[r] + rs;
      mrun[r] = mnew;
    }
#pragma unroll
    for (int t = 0; t < 4; ++t)
#pragma unroll
      for (int r = 0; r < 8; ++r) O[t][r] *= scl[r];

    // P: C-layout regs -> LDS -> A-layout regs (wave-private; DS in-order)
#pragma unroll
    for (int r = 0; r < 8; ++r) {
      int m = r + 8 * hi;
      P[m * 32 + ln]      = (__bf16)s0[r];
      P[m * 32 + 16 + ln] = (__bf16)s1[r];
    }
    __builtin_amdgcn_wave_barrier();
    bf16x16 pa;
#pragma unroll
    for (int p = 0; p < 8; ++p) {
      int kbase = ((2 * p) & 7) + ((p & 4) ? 16 : 0) + hi * 8;
      pa[2 * p]     = P[ln * 32 + kbase];
      pa[2 * p + 1] = P[ln * 32 + kbase + 1];
    }
    __builtin_amdgcn_wave_barrier();

#pragma unroll
    for (int dt = 0; dt < 4; ++dt)
      O[dt] = wmma_bf16(pa, Vbase[(size_t)(dt * 64 + kv) * 32], O[dt]);
  }

  // normalize and scatter into ctx A-panels [mtile][kstep(32)][lane][16]
  float linv[8];
#pragma unroll
  for (int r = 0; r < 8; ++r) linv[r] = 1.0f / lrun[r];

  int b = bh >> 4, h = bh & 15;
  int mtile = b * 128 + qtile;
#pragma unroll
  for (int dt = 0; dt < 4; ++dt) {
#pragma unroll
    for (int r = 0; r < 8; ++r) {
      float v = O[dt][r] * linv[r];
      int s    = qtile * 16 + r + 8 * hi;
      int c    = h * 64 + dt * 16 + ln;
      int kst  = c >> 5, koff = c & 31;
      int laneA = (s & 15) + 16 * ((koff >> 3) & 1);
      int iA    = (koff & 7) + ((koff & 16) ? 8 : 0);
      ctxA[(((size_t)mtile * 32 + kst) * 32 + laneA) * 16 + iA] = (__bf16)v;
    }
  }
}

// ---------------------------------------------------------------------------
extern "C" void kernel_launch(void* const* d_in, const int* in_sizes, int n_in,
                              void* d_out, int out_size, void* d_ws,
                              size_t ws_size, hipStream_t stream) {
  const float* x  = (const float*)d_in[0];
  const float* Wq = (const float*)d_in[1];
  const float* bq = (const float*)d_in[2];
  const float* Wk = (const float*)d_in[3];
  const float* bk = (const float*)d_in[4];
  const float* Wv = (const float*)d_in[5];
  const float* bv = (const float*)d_in[6];
  const float* Wo = (const float*)d_in[7];
  const float* bo = (const float*)d_in[8];

  char* ws = (char*)d_ws;
  const size_t MB = 1u << 20;
  __bf16* xA   = (__bf16*)(ws + 0 * MB);   // 8 MB  x A-panels
  __bf16* WqB  = (__bf16*)(ws + 8 * MB);   // 2 MB
  __bf16* WkB  = (__bf16*)(ws + 10 * MB);  // 2 MB
  __bf16* WvB  = (__bf16*)(ws + 12 * MB);  // 2 MB
  __bf16* WoB  = (__bf16*)(ws + 14 * MB);  // 2 MB
  __bf16* qAp  = (__bf16*)(ws + 16 * MB);  // 8 MB  Q A-panels
  __bf16* kBp  = (__bf16*)(ws + 24 * MB);  // 8 MB  K B-panels
  __bf16* vBp  = (__bf16*)(ws + 32 * MB);  // 8 MB  V B-panels
  __bf16* ctxA = (__bf16*)(ws + 40 * MB);  // 8 MB  ctx A-panels

  // 1) pack inputs into WMMA panel layouts (bf16)
  pack_A_f32<<<(kRows * kHid) / 256, 256, 0, stream>>>(x, xA, kHid);
  pack_B_f32<<<(kHid * kHid) / 256, 256, 0, stream>>>(Wq, WqB, kHid, kHid);
  pack_B_f32<<<(kHid * kHid) / 256, 256, 0, stream>>>(Wk, WkB, kHid, kHid);
  pack_B_f32<<<(kHid * kHid) / 256, 256, 0, stream>>>(Wv, WvB, kHid, kHid);
  pack_B_f32<<<(kHid * kHid) / 256, 256, 0, stream>>>(Wo, WoB, kHid, kHid);

  // 2) QKV projections (WMMA), scattering straight into attention panels
  int gemm_blocks = (kMT / 2) * (kNT / 4) / 8;   // 256 blocks of 8 waves
  gemm_wmma<<<gemm_blocks, 256, 0, stream>>>(xA, WqB, bq, qAp, 0);
  gemm_wmma<<<gemm_blocks, 256, 0, stream>>>(xA, WkB, bk, kBp, 1);
  gemm_wmma<<<gemm_blocks, 256, 0, stream>>>(xA, WvB, bv, vBp, 2);

  // 3) fused flash attention: grid (128 qtiles / 8 waves, 32 bh)
  flash_attn<<<dim3(16, 32), 256, 0, stream>>>(qAp, kBp, vBp, ctxA);

  // 4) output projection -> fp32 d_out
  gemm_wmma<<<gemm_blocks, 256, 0, stream>>>(ctxA, WoB, bo, d_out, 3);
}
